// ClassAttn_6322191859907
// MI455X (gfx1250) — matrix operand order repaired and checked by reading, not compiled, not run on hardware
//
#include <hip/hip_runtime.h>
#include <hip/hip_bf16.h>

// Problem constants (match reference)
#define B_  32
#define N_  4096
#define C_  1024
#define H_  16
#define HD_ 16
#define CR_ 256

#define SLICES_   16     // workgroups per batch in fused kernel
#define TOKS_WG_  256    // tokens per workgroup
#define CHUNK_    32     // tokens per chunk (= WMMA K for pass 2)
#define XST_STR_  40     // xsT row stride in bf16 elems (80B: 16B aligned, bank-spread)

typedef __attribute__((ext_vector_type(16))) __bf16 v16bf;
typedef __attribute__((ext_vector_type(8)))  float  v8f;

union Frag {
    uint4    q[2];
    unsigned u[8];
    v16bf    v;
};

// round-to-nearest-even f32 -> bf16, packed pair
__device__ __forceinline__ unsigned pack2bf(float a, float b) {
    unsigned ua = __builtin_bit_cast(unsigned, a);
    unsigned ub = __builtin_bit_cast(unsigned, b);
    ua += 0x7FFFu + ((ua >> 16) & 1u);
    ub += 0x7FFFu + ((ub >> 16) & 1u);
    return (ua >> 16) | (ub & 0xFFFF0000u);
}
__device__ __forceinline__ unsigned short bf16of(float a) {
    unsigned ua = __builtin_bit_cast(unsigned, a);
    ua += 0x7FFFu + ((ua >> 16) & 1u);
    return (unsigned short)(ua >> 16);
}

// ---------------------------------------------------------------------------
// K1: q[b,h,:] = (x[b,0,:] @ Wq_h) * SCALE;  wkq[b,c,h] = Wk[c,h*16:+16] . q[b,h,:]
//     wkq stored bf16, pre-packed in WMMA B-fragment order:
//     (b, kc=c/32, lane = h + 16*((c>>3)&1), slot = (c&7) + 8*(c%32>=16))
// ---------------------------------------------------------------------------
__global__ void __launch_bounds__(256)
k1_q_wkq(const float* __restrict__ x, const float* __restrict__ Wq,
         const float* __restrict__ Wk, unsigned short* __restrict__ wkq)
{
    __shared__ float qs[CR_];
    const int b = blockIdx.x;
    const int t = threadIdx.x;

    const float* xrow = x + (size_t)b * N_ * C_;   // class token row
    float acc = 0.f;
    for (int c = 0; c < C_; ++c)
        acc += xrow[c] * Wq[(size_t)c * CR_ + t];
    qs[t] = acc * 0.25f;                            // SCALE = 16^-0.5
    __syncthreads();

    for (int i = 0; i < 64; ++i) {
        int p = t + i * 256;            // p in [0, 16384)
        int c = p >> 4, h = p & 15;
        const float* wkrow = Wk + (size_t)c * CR_ + h * HD_;
        float s = 0.f;
        #pragma unroll
        for (int d = 0; d < HD_; ++d) s += wkrow[d] * qs[h * HD_ + d];
        int kc   = c >> 5;
        int kl   = c & 31;
        int lane = h + ((kl & 8) ? 16 : 0);
        int slot = (kl & 7) + ((kl & 16) ? 8 : 0);
        wkq[(((size_t)b * 32 + kc) * 32 + lane) * 16 + slot] = bf16of(s);
    }
}

// ---------------------------------------------------------------------------
// Fused flash-decode pass: per WG = (batch, 256-token slice).
// Single HBM read of x; logits WMMA + online softmax + value WMMA per chunk.
// Emits per-slice partial accumulators (unnormalized) + running (m, l).
// ---------------------------------------------------------------------------
__global__ void __launch_bounds__(256)
k_fused(const float* __restrict__ x, const unsigned short* __restrict__ wkq,
        float* __restrict__ pacc, float* __restrict__ pm, float* __restrict__ pl)
{
    __shared__ __align__(16) unsigned short xsT[C_ * XST_STR_];  // 80 KiB [ch][tok] bf16
    __shared__ __align__(16) float part[8 * 32 * 8];             // 8 KiB  [wave][lane][r]
    __shared__ __align__(16) float lbuf[CHUNK_ * H_];            // 2 KiB  [tokL][head]
    __shared__ __align__(16) unsigned short bfrag[32 * 16];      // 1 KiB  prob B-fragment
    __shared__ float mrun[H_], lrun[H_], scaleB[H_];

    const int b     = blockIdx.x >> 4;
    const int slice = blockIdx.x & 15;
    const int tid   = threadIdx.x;
    const int wave  = tid >> 5;
    const int lane  = tid & 31;
    const int ln15  = lane & 15;
    const int khalf = lane >> 4;
    const int koff  = khalf * 8;           // 16-bit A-frag K offset

    if (tid < H_) { mrun[tid] = -1e30f; lrun[tid] = 0.f; }

    const int tileP1  = wave & 1;          // 16-token tile within chunk
    const int kslice  = wave >> 1;         // K slice (256 channels)

    v8f acc[8];
    #pragma unroll
    for (int i = 0; i < 8; ++i) acc[i] = (v8f){};

    const size_t xbase = (size_t)b * N_ * C_;
    const uint4* wkqb  = (const uint4*)wkq + ((size_t)b * 1024 + lane) * 2;

    __syncthreads();

    for (int chunk = 0; chunk < TOKS_WG_ / CHUNK_; ++chunk) {
        const int tok0 = slice * TOKS_WG_ + chunk * CHUNK_;

        // ---- P1: K-split partial logits; stage bf16 x^T into LDS ----
        {
            const int  tokL = tileP1 * 16 + ln15;              // token within chunk
            const int  m    = tok0 + tokL;                     // global token
            const float* arow = x + xbase + (size_t)m * C_;
            v8f lacc = {};
            for (int ks = 0; ks < 8; ++ks) {
                const int kc = kslice * 8 + ks;                // 32-channel K step
                const float* ap = arow + kc * 32 + koff;
                float4 f0 = *(const float4*)(ap);
                float4 f1 = *(const float4*)(ap + 4);
                float4 f2 = *(const float4*)(ap + 16);
                float4 f3 = *(const float4*)(ap + 20);
                Frag A;
                A.u[0]=pack2bf(f0.x,f0.y); A.u[1]=pack2bf(f0.z,f0.w);
                A.u[2]=pack2bf(f1.x,f1.y); A.u[3]=pack2bf(f1.z,f1.w);
                A.u[4]=pack2bf(f2.x,f2.y); A.u[5]=pack2bf(f2.z,f2.w);
                A.u[6]=pack2bf(f3.x,f3.y); A.u[7]=pack2bf(f3.z,f3.w);

                // transposed bf16 staging for pass 2 (x fetched from HBM once)
                const int chb = kc * 32 + koff;
                #pragma unroll
                for (int j = 0; j < 4; ++j) {
                    xsT[(chb + 2*j    ) * XST_STR_ + tokL] = (unsigned short)(A.u[j]);
                    xsT[(chb + 2*j + 1) * XST_STR_ + tokL] = (unsigned short)(A.u[j] >> 16);
                }
                #pragma unroll
                for (int j = 0; j < 4; ++j) {
                    xsT[(chb + 16 + 2*j    ) * XST_STR_ + tokL] = (unsigned short)(A.u[4+j]);
                    xsT[(chb + 16 + 2*j + 1) * XST_STR_ + tokL] = (unsigned short)(A.u[4+j] >> 16);
                }

                Frag Bm;
                const uint4* bp = wkqb + (size_t)kc * 64;
                Bm.q[0] = bp[0]; Bm.q[1] = bp[1];
                __builtin_prefetch(ap + 32, 0, 1);

                lacc = __builtin_amdgcn_wmma_f32_16x16x32_bf16(
                           false, A.v, false, Bm.v, (short)0, lacc, false, false);
            }
            float* pp = part + ((size_t)wave * 32 + lane) * 8;
            #pragma unroll
            for (int r = 0; r < 8; ++r) pp[r] = lacc[r];
        }
        __syncthreads();

        // ---- reduce the 4 K-slice partials -> lbuf[tokL][head] ----
        #pragma unroll
        for (int i = 0; i < 2; ++i) {
            int e  = tid + i * 256;               // 0..511
            int mg = e >> 4, n = e & 15;          // tokL, head
            int tt = mg >> 4, mm = mg & 15;
            int pln = n + ((mm & 8) ? 16 : 0);
            int r   = mm & 7;
            float s = 0.f;
            #pragma unroll
            for (int ks = 0; ks < 4; ++ks)
                s += part[((tt + 2*ks) * 32 + pln) * 8 + r];
            lbuf[mg * 16 + n] = s;
        }
        __syncthreads();

        // ---- online softmax update (per head) ----
        if (tid < H_) {
            const int h = tid;
            float mnew = mrun[h];
            for (int tk = 0; tk < CHUNK_; ++tk) mnew = fmaxf(mnew, lbuf[tk*16 + h]);
            float fac = __expf(mrun[h] - mnew);
            float sum = 0.f;
            for (int tk = 0; tk < CHUNK_; ++tk) {
                float p = __expf(lbuf[tk*16 + h] - mnew);
                sum += p;
                // B-fragment order: lane = h + 16*((tk>>3)&1), slot = (tk&7) + 8*(tk>=16)
                bfrag[(h + ((tk & 8) ? 16 : 0)) * 16 + (tk & 7) + ((tk & 16) ? 8 : 0)] = bf16of(p);
            }
            lrun[h] = lrun[h] * fac + sum;
            mrun[h] = mnew;
            scaleB[h] = fac;
        }
        __syncthreads();

        // ---- P2: rescale accs + accumulate prob-weighted x (K = 32 tokens) ----
        {
            const float fac = scaleB[ln15];        // each lane's accs share one head
            Frag Bm;
            const uint4* bq = (const uint4*)bfrag + lane * 2;
            Bm.q[0] = bq[0]; Bm.q[1] = bq[1];
            #pragma unroll
            for (int ct = 0; ct < 8; ++ct) {
                const int ch = (wave * 8 + ct) * 16 + ln15;   // A-frag M row = channel
                Frag A;
                A.q[0] = *(const uint4*)(xsT + ch * XST_STR_ + koff);
                A.q[1] = *(const uint4*)(xsT + ch * XST_STR_ + koff + 16);
                v8f a = acc[ct];
                #pragma unroll
                for (int r = 0; r < 8; ++r) a[r] *= fac;
                acc[ct] = __builtin_amdgcn_wmma_f32_16x16x32_bf16(
                              false, A.v, false, Bm.v, (short)0, a, false, false);
            }
        }
        __syncthreads();   // protect xsT/part/bfrag before next chunk
    }

    // ---- epilogue: per-slice partials to global ----
    const int bs = b * SLICES_ + slice;
    #pragma unroll
    for (int ct = 0; ct < 8; ++ct) {
        #pragma unroll
        for (int r = 0; r < 8; ++r) {
            int ch = (wave * 8 + ct) * 16 + r + khalf * 8;    // D: M = r + 8*(lane>=16)
            pacc[(((size_t)bs) * H_ + ln15) * C_ + ch] = acc[ct][r];
        }
    }
    if (tid < H_) {
        pm[(size_t)bs * H_ + tid] = mrun[tid];
        pl[(size_t)bs * H_ + tid] = lrun[tid];
    }
}

// ---------------------------------------------------------------------------
// Combine slice partials: xattn[b,h,c] = sum_s w_s[h]*pacc[s,h,c],
// w_s = exp(m_s - M) / sum_s l_s * exp(m_s - M)
// ---------------------------------------------------------------------------
__global__ void __launch_bounds__(256)
k_combine(const float* __restrict__ pacc, const float* __restrict__ pm,
          const float* __restrict__ pl, float* __restrict__ xattn)
{
    __shared__ float wsc[SLICES_ * H_];
    const int b = blockIdx.x, t = threadIdx.x;
    if (t < H_) {
        const int h = t;
        float M = -1e30f;
        for (int s = 0; s < SLICES_; ++s)
            M = fmaxf(M, pm[((size_t)b * SLICES_ + s) * H_ + h]);
        float denom = 0.f;
        for (int s = 0; s < SLICES_; ++s)
            denom += pl[((size_t)b * SLICES_ + s) * H_ + h] *
                     __expf(pm[((size_t)b * SLICES_ + s) * H_ + h] - M);
        float inv = 1.0f / denom;
        for (int s = 0; s < SLICES_; ++s)
            wsc[s * H_ + h] = __expf(pm[((size_t)b * SLICES_ + s) * H_ + h] - M) * inv;
    }
    __syncthreads();
    for (int i = 0; i < 64; ++i) {
        int e = t + i * 256;            // 0..16383
        int h = e >> 10, c = e & 1023;
        float s = 0.f;
        #pragma unroll 4
        for (int sl = 0; sl < SLICES_; ++sl)
            s += pacc[(((size_t)(b * SLICES_ + sl)) * H_ + h) * C_ + c] * wsc[sl * H_ + h];
        xattn[((size_t)b * H_ + h) * C_ + c] = s;
    }
}

// ---------------------------------------------------------------------------
// K5a: o[b, h*16+d] = sum_c xattn[b,h,c] * Wv[c, h*16+d]   (tiny)
// ---------------------------------------------------------------------------
__global__ void __launch_bounds__(256)
k5a_o(const float* __restrict__ xattn, const float* __restrict__ Wv,
      float* __restrict__ o)
{
    __shared__ float red[256];
    const int bh = blockIdx.x;
    const int h  = bh & 15;
    const int t  = threadIdx.x;
    const int d  = t & 15, seg = t >> 4;

    const float* xa = xattn + (size_t)bh * C_;
    float acc = 0.f;
    for (int i = 0; i < 64; ++i) {
        int c = seg * 64 + i;
        acc += xa[c] * Wv[(size_t)c * CR_ + h * HD_ + d];
    }
    red[t] = acc; __syncthreads();
    if (seg == 0) {
        float s = 0.f;
        #pragma unroll
        for (int ss = 0; ss < 16; ++ss) s += red[ss * 16 + d];
        o[(size_t)(bh >> 4) * CR_ + h * HD_ + d] = s;
    }
}

// ---------------------------------------------------------------------------
// K5b: out[b, :] = o[b, :] @ Wp + bp   (tiny)
// ---------------------------------------------------------------------------
__global__ void __launch_bounds__(256)
k5b_proj(const float* __restrict__ o, const float* __restrict__ Wp,
         const float* __restrict__ bp, float* __restrict__ out)
{
    __shared__ float os[CR_];
    const int b = blockIdx.x, t = threadIdx.x;
    os[t] = o[(size_t)b * CR_ + t];
    __syncthreads();

    const int j = t * 4;
    float4 acc = *(const float4*)(bp + j);
    for (int r = 0; r < CR_; ++r) {
        float  ov = os[r];
        float4 w  = *(const float4*)(Wp + (size_t)r * C_ + j);
        acc.x += ov * w.x; acc.y += ov * w.y;
        acc.z += ov * w.z; acc.w += ov * w.w;
    }
    *(float4*)(out + (size_t)b * C_ + j) = acc;
}

// ---------------------------------------------------------------------------
extern "C" void kernel_launch(void* const* d_in, const int* in_sizes, int n_in,
                              void* d_out, int out_size, void* d_ws, size_t ws_size,
                              hipStream_t stream)
{
    const float* x  = (const float*)d_in[0];
    const float* Wq = (const float*)d_in[1];
    const float* Wk = (const float*)d_in[2];
    const float* Wv = (const float*)d_in[3];
    const float* Wp = (const float*)d_in[4];
    const float* bp = (const float*)d_in[5];
    float* out = (float*)d_out;

    char* ws = (char*)d_ws;
    unsigned short* wkq = (unsigned short*)(ws);            // 1 MiB  wkq B-fragments
    float* pacc  = (float*)(ws + ((size_t)1  << 20));       // 32 MiB [B*16][H][C] partial
    float* pm    = (float*)(ws + ((size_t)33 << 20));       // 32 KiB running max
    float* pl    = (float*)(ws + ((size_t)33 << 20) + (1u << 16)); // 32 KiB running sum
    float* xattn = (float*)(ws + ((size_t)34 << 20));       // 2 MiB  [B][H][C]
    float* o     = (float*)(ws + ((size_t)36 << 20));       // 32 KiB [B][CR]

    k1_q_wkq <<<B_,            256, 0, stream>>>(x, Wq, Wk, wkq);
    k_fused  <<<B_ * SLICES_,  256, 0, stream>>>(x, wkq, pacc, pm, pl);
    k_combine<<<B_,            256, 0, stream>>>(pacc, pm, pl, xattn);
    k5a_o    <<<B_ * H_,       256, 0, stream>>>(xattn, Wv, o);
    k5b_proj <<<B_,            256, 0, stream>>>(o, Wp, bp, out);
}
